// HetSAGE_3401614098572
// MI455X (gfx1250) — compile-verified, matrix operand order checked by reading, not run on hardware
//
#include <hip/hip_runtime.h>

// ---------------------------------------------------------------------------
// HetSAGE on MI455X (gfx1250): bf16 WMMA GEMMs + L2-resident atomic-max scatter
// ---------------------------------------------------------------------------

typedef __attribute__((ext_vector_type(16))) __bf16 v16bf;
typedef __attribute__((ext_vector_type(8)))  float  v8f;

#define D_HID      256   // hidden dim (all GEMM outputs are [*,256])
#define TILE_M     32    // output rows per workgroup
#define KSTEP      32    // K per WMMA (bf16 16x16x32)
#define LDS_STRIDE 40    // shorts per LDS row (80B, conflict-free for b128 reads)
#define SHC_STRIDE (D_HID + 4)

__device__ __forceinline__ unsigned short f2bf(float f) {
    unsigned int u = __float_as_uint(f);
    u += 0x7FFFu + ((u >> 16) & 1u);        // round-to-nearest-even
    return (unsigned short)(u >> 16);
}

union Frag { v16bf v; uint4 q[2]; };

// C[M,256] = act(A[M,K1] @ W[256,K1]^T  (+ A2[M,K2] @ W2[256,K2]^T) + bias)
// optionally layernorm(gamma,beta) per row, then write or accumulate into Cout.
__global__ __launch_bounds__(256)
void gemm_bias_act_ln(const float* __restrict__ A,  const float* __restrict__ W,  int K1,
                      const float* __restrict__ A2, const float* __restrict__ W2, int K2,
                      const float* __restrict__ bias,
                      const float* __restrict__ gamma, const float* __restrict__ beta,
                      float* __restrict__ Cout,
                      int do_relu, int do_ln, int do_accum)
{
    __shared__ unsigned short shA[TILE_M * LDS_STRIDE];   // 32 x 32 bf16 tile (padded)
    __shared__ unsigned short shB[D_HID * LDS_STRIDE];    // 256 x 32 bf16 tile (padded)
    __shared__ float          shC[TILE_M * SHC_STRIDE];   // epilogue staging for layernorm

    const int tid      = threadIdx.x;
    const int lane     = tid & 31;
    const int wid      = tid >> 5;
    const int rowHalf  = wid >> 2;       // 0..1 -> which 16-row half of the 32-row tile
    const int colGroup = wid & 3;        // 0..3 -> 64-column group
    const int hlf      = lane >> 4;      // 0..1 (K-chunk selector per ISA layout)
    const int l15      = lane & 15;
    const int rowBase  = blockIdx.x * TILE_M;

    v8f acc[4];
    #pragma unroll
    for (int j = 0; j < 4; ++j) {
        #pragma unroll
        for (int i = 0; i < 8; ++i) acc[j][i] = 0.0f;
    }

    for (int s = 0; s < 2; ++s) {
        const float* Ap = s ? A2 : A;
        const float* Wp = s ? W2 : W;
        const int    Ks = s ? K2 : K1;
        if (Ks == 0) continue;

        for (int k0 = 0; k0 < Ks; k0 += KSTEP) {
            // ---- stage A tile: 32 rows x 32 K, fp32 -> bf16 ----
            {
                int row = tid >> 3;
                int c4  = (tid & 7) * 4;
                float4 v = *(const float4*)(Ap + (size_t)(rowBase + row) * Ks + k0 + c4);
                unsigned int p0 = (unsigned int)f2bf(v.x) | ((unsigned int)f2bf(v.y) << 16);
                unsigned int p1 = (unsigned int)f2bf(v.z) | ((unsigned int)f2bf(v.w) << 16);
                *(uint2*)(shA + row * LDS_STRIDE + c4) = make_uint2(p0, p1);
            }
            // ---- stage B tile: 256 W-rows x 32 K, fp32 -> bf16 ----
            #pragma unroll
            for (int i = 0; i < 8; ++i) {
                int chunk = tid + i * 256;          // 0..2047 float4 chunks
                int row   = chunk >> 3;             // 0..255
                int c4    = (chunk & 7) * 4;
                float4 v = *(const float4*)(Wp + (size_t)row * Ks + k0 + c4);
                unsigned int p0 = (unsigned int)f2bf(v.x) | ((unsigned int)f2bf(v.y) << 16);
                unsigned int p1 = (unsigned int)f2bf(v.z) | ((unsigned int)f2bf(v.w) << 16);
                *(uint2*)(shB + row * LDS_STRIDE + c4) = make_uint2(p0, p1);
            }
            __syncthreads();

            // ---- load A fragment and ALL 4 B fragments first (batched ds_load_b128,
            //      one wait), then issue the 4 WMMAs back-to-back so the matrix pipe
            //      and LDS pipe can overlap ----
            Frag a;
            {
                int offA = (rowHalf * 16 + l15) * LDS_STRIDE + hlf * 8;
                a.q[0] = *(const uint4*)(shA + offA);
                a.q[1] = *(const uint4*)(shA + offA + 16);
            }
            Frag b[4];
            #pragma unroll
            for (int j = 0; j < 4; ++j) {
                int n    = colGroup * 64 + j * 16 + l15;
                int offB = n * LDS_STRIDE + hlf * 8;
                b[j].q[0] = *(const uint4*)(shB + offB);
                b[j].q[1] = *(const uint4*)(shB + offB + 16);
            }
            #pragma unroll
            for (int j = 0; j < 4; ++j) {
                acc[j] = __builtin_amdgcn_wmma_f32_16x16x32_bf16(
                    false, a.v, false, b[j].v, (short)0, acc[j], false, false);
            }
            __syncthreads();
        }
    }

    // ---- epilogue: bias (+relu), then direct store or LDS-staged layernorm ----
    #pragma unroll
    for (int j = 0; j < 4; ++j) {
        int   col = colGroup * 64 + j * 16 + l15;
        float bv  = bias[col];
        #pragma unroll
        for (int v = 0; v < 8; ++v) {
            int   mrow = rowHalf * 16 + v + hlf * 8;   // C/D layout: lanes16-31 -> M=v+8
            float val  = acc[j][v] + bv;
            if (do_relu) val = fmaxf(val, 0.0f);
            if (do_ln) shC[mrow * SHC_STRIDE + col] = val;
            else       Cout[(size_t)(rowBase + mrow) * D_HID + col] = val;
        }
    }

    if (do_ln) {
        __syncthreads();
        int r   = tid >> 3;      // 32 rows
        int seg = tid & 7;       // 8 threads per row, 32 cols each
        const float* rowp = shC + r * SHC_STRIDE;
        float s = 0.0f, q = 0.0f;
        #pragma unroll
        for (int i = 0; i < 32; ++i) {
            float x = rowp[seg * 32 + i];
            s += x; q += x * x;
        }
        #pragma unroll
        for (int off = 1; off < 8; off <<= 1) {
            s += __shfl_xor(s, off, 8);
            q += __shfl_xor(q, off, 8);
        }
        float mean = s * (1.0f / 256.0f);
        float var  = q * (1.0f / 256.0f) - mean * mean;
        float rstd = rsqrtf(var + 1e-5f);
        float* drow = Cout + (size_t)(rowBase + r) * D_HID;
        #pragma unroll
        for (int i = 0; i < 32; ++i) {
            int   c = seg * 32 + i;
            float y = (rowp[c] - mean) * rstd * gamma[c] + beta[c];
            if (do_accum) drow[c] += y;
            else          drow[c]  = y;
        }
    }
}

// neigh[dst[e],:] = max(neigh[dst[e],:], hp[src[e],:])  via u32 atomic max on
// nonnegative float bits (hp is post-relu, neigh initialized to 0.0f == 0u).
__global__ __launch_bounds__(256)
void scatter_max_kernel(const float* __restrict__ hp,
                        const int* __restrict__ src,
                        const int* __restrict__ dst,
                        unsigned int* __restrict__ neigh,
                        int E)
{
    int e = blockIdx.x * 8 + (threadIdx.x >> 5);   // one wave per edge
    if (e >= E) return;
    int lane = threadIdx.x & 31;
    int s = src[e];
    int d = dst[e];
    const float*  srow = hp + (size_t)s * D_HID;
    unsigned int* drow = neigh + (size_t)d * D_HID;
    #pragma unroll
    for (int i = 0; i < 2; ++i) {
        int c = i * 128 + lane * 4;
        float4 v = *(const float4*)(srow + c);
        if (v.x > 0.0f) atomicMax(drow + c,     __float_as_uint(v.x));
        if (v.y > 0.0f) atomicMax(drow + c + 1, __float_as_uint(v.y));
        if (v.z > 0.0f) atomicMax(drow + c + 2, __float_as_uint(v.z));
        if (v.w > 0.0f) atomicMax(drow + c + 3, __float_as_uint(v.w));
    }
}

__global__ __launch_bounds__(256)
void zero_kernel(uint4* __restrict__ p, int n4)
{
    int i = blockIdx.x * 256 + threadIdx.x;
    if (i < n4) p[i] = make_uint4(0u, 0u, 0u, 0u);
}

// ---------------------------------------------------------------------------

extern "C" void kernel_launch(void* const* d_in, const int* in_sizes, int n_in,
                              void* d_out, int out_size, void* d_ws, size_t ws_size,
                              hipStream_t stream)
{
    const float* x      = (const float*)d_in[0];
    const int*   src    = (const int*)  d_in[1];
    const int*   dst    = (const int*)  d_in[2];
    const float* Wlin   = (const float*)d_in[3];
    const float* blin   = (const float*)d_in[4];
    const float* Wpool  = (const float*)d_in[5];
    const float* bpool  = (const float*)d_in[6];
    const float* Wself  = (const float*)d_in[7];
    const float* Wneigh = (const float*)d_in[8];
    const float* bconv  = (const float*)d_in[9];
    const float* gammap = (const float*)d_in[10];
    const float* betap  = (const float*)d_in[11];

    const int D = D_HID, L = 3, T = 2;
    const int F = in_sizes[3] / D;       // 512
    const int N = in_sizes[0] / F;       // 20000
    const int E = in_sizes[1] / T;       // 200000

    const size_t nd = (size_t)N * D;
    float* hbuf  = (float*)d_ws;
    float* obuf  = hbuf + nd;
    float* hpbuf = hbuf + 2 * nd;
    float* neigh = hbuf + 3 * nd;

    dim3 blk(256);
    int gemmGrid = (N + TILE_M - 1) / TILE_M;
    int n4       = (int)(nd / 4);
    int zeroGrid = (n4 + 255) / 256;
    int scatGrid = (E + 7) / 8;

    // HeteroLinear: h = x @ Wlin^T + blin
    gemm_bias_act_ln<<<gemmGrid, blk, 0, stream>>>(
        x, Wlin, F, nullptr, nullptr, 0,
        blin, nullptr, nullptr, hbuf, /*relu*/0, /*ln*/0, /*accum*/0);

    const float* hin = hbuf;
    for (int l = 0; l < L; ++l) {
        float* dest = (l == L - 1) ? (float*)d_out : ((hin == hbuf) ? obuf : hbuf);
        for (int t = 0; t < T; ++t) {
            size_t wo = ((size_t)l * T + t) * (size_t)D * D;
            size_t bo = ((size_t)l * T + t) * (size_t)D;

            // hp = relu(h @ Wpool^T + bpool)
            gemm_bias_act_ln<<<gemmGrid, blk, 0, stream>>>(
                hin, Wpool + wo, D, nullptr, nullptr, 0,
                bpool + bo, nullptr, nullptr, hpbuf, 1, 0, 0);

            // neigh = 0; segment-max over edges
            zero_kernel<<<zeroGrid, blk, 0, stream>>>((uint4*)neigh, n4);
            scatter_max_kernel<<<scatGrid, blk, 0, stream>>>(
                hpbuf, src + (size_t)t * E, dst + (size_t)t * E,
                (unsigned int*)neigh, E);

            // out (+)= layernorm( act( h@Wself^T + neigh@Wneigh^T + bconv ) )
            gemm_bias_act_ln<<<gemmGrid, blk, 0, stream>>>(
                hin, Wself + wo, D, neigh, Wneigh + wo, D,
                bconv + bo, gammap + bo, betap + bo, dest,
                /*relu*/(l < L - 1) ? 1 : 0, /*ln*/1, /*accum*/t);
        }
        hin = dest;
    }
    (void)n_in; (void)out_size; (void)ws_size;
}
